// mynet_77592879170081
// MI455X (gfx1250) — compile-verified, matrix operand order checked by reading, not run on hardware
//
#include <hip/hip_runtime.h>
#include <math.h>

typedef __attribute__((ext_vector_type(16))) _Float16 v16h;
typedef __attribute__((ext_vector_type(8)))  _Float16 v8h;
typedef __attribute__((ext_vector_type(8)))  float    v8f;

// Load one 16-lane-wide f16 fragment half-pair: 8 contiguous halves at p and
// 8 at p+16 (two global_load_b128), merged into the v16h WMMA operand.
__device__ __forceinline__ v16h ldfrag(const _Float16* __restrict__ p)
{
    v8h lo = *(const v8h*)p;
    v8h hi = *(const v8h*)(p + 16);
    return __builtin_shufflevector(lo, hi, 0, 1, 2, 3, 4, 5, 6, 7,
                                           8, 9, 10, 11, 12, 13, 14, 15);
}

// ---------------------------------------------------------------------------
// WMMA GEMM on pre-padded f16 operands, software-pipelined k-loop.
//   A : [M, lda] f16, rows dense, cols [K..Kp) zero-padded, M % 16 == 0
//   Bt: [Np, Kp] f16 = transpose(W) zero-padded, Np % 64 == 0, Kp % 32 == 0
//   Y : f32 (Yf) or f16 (Yh) output, row stride ldy; optional bias + tanh.
// One wave computes a 16x64 strip (A fragment reused across 4 v_wmma).
// Next k-step fragments are loaded into fresh registers before the current
// step's WMMAs issue, so loads overlap matrix math (partial loadcnt waits).
// ---------------------------------------------------------------------------
__global__ void wmma_gemm_kernel(const _Float16* __restrict__ A, int lda,
                                 const _Float16* __restrict__ Bt, int Kp,
                                 const float* __restrict__ bias,
                                 float* __restrict__ Yf,
                                 _Float16* __restrict__ Yh, int ldy,
                                 int M, int N, int act)
{
    const int lane  = threadIdx.x;                    // 0..31
    const int tileM = (blockIdx.y * blockDim.y + threadIdx.y) * 16;
    if (tileM >= M) return;                           // uniform per wave
    const int tileN = blockIdx.x * 64;

    const bool hi  = lane >= 16;
    const int  l15 = lane & 15;

    const _Float16* ap = A  + (long)(tileM + l15) * lda + (hi ? 8 : 0);
    const _Float16* bp = Bt + (long)(tileN + l15) * Kp  + (hi ? 8 : 0);
    const long bs = (long)16 * Kp;

    v8f c0 = {}, c1 = {}, c2 = {}, c3 = {};

    // prologue: first k-step fragments
    v16h a  = ldfrag(ap);
    v16h b0 = ldfrag(bp);
    v16h b1 = ldfrag(bp + bs);
    v16h b2 = ldfrag(bp + 2 * bs);
    v16h b3 = ldfrag(bp + 3 * bs);

    for (int k0 = 32; k0 < Kp; k0 += 32) {
        ap += 32;
        bp += 32;
        // issue next-step loads before consuming current fragments
        v16h an  = ldfrag(ap);
        v16h bn0 = ldfrag(bp);
        v16h bn1 = ldfrag(bp + bs);
        v16h bn2 = ldfrag(bp + 2 * bs);
        v16h bn3 = ldfrag(bp + 3 * bs);
        c0 = __builtin_amdgcn_wmma_f32_16x16x32_f16(false, a, false, b0, (short)0, c0, false, false);
        c1 = __builtin_amdgcn_wmma_f32_16x16x32_f16(false, a, false, b1, (short)0, c1, false, false);
        c2 = __builtin_amdgcn_wmma_f32_16x16x32_f16(false, a, false, b2, (short)0, c2, false, false);
        c3 = __builtin_amdgcn_wmma_f32_16x16x32_f16(false, a, false, b3, (short)0, c3, false, false);
        a = an; b0 = bn0; b1 = bn1; b2 = bn2; b3 = bn3;
    }
    // epilogue k-step
    c0 = __builtin_amdgcn_wmma_f32_16x16x32_f16(false, a, false, b0, (short)0, c0, false, false);
    c1 = __builtin_amdgcn_wmma_f32_16x16x32_f16(false, a, false, b1, (short)0, c1, false, false);
    c2 = __builtin_amdgcn_wmma_f32_16x16x32_f16(false, a, false, b2, (short)0, c2, false, false);
    c3 = __builtin_amdgcn_wmma_f32_16x16x32_f16(false, a, false, b3, (short)0, c3, false, false);

    const int rbase = tileM + (hi ? 8 : 0);
#pragma unroll
    for (int j = 0; j < 4; ++j) {
        const int bcol = tileN + j * 16 + l15;
        if (bcol >= N) continue;
        const float bv = bias ? bias[bcol] : 0.0f;
        const v8f cc = (j == 0) ? c0 : (j == 1) ? c1 : (j == 2) ? c2 : c3;
#pragma unroll
        for (int r = 0; r < 8; ++r) {
            float v = cc[r] + bv;
            if (act == 1) v = tanhf(v);
            const long o = (long)(rbase + r) * ldy + bcol;
            if (Yh) Yh[o] = (_Float16)v;
            else    Yf[o] = v;
        }
    }
}

// Row-wise LayerNorm (f32 in, f16 out with stride), one wave32 per row.
__global__ void layernorm_f16_kernel(const float* __restrict__ X, int ldx,
                                     const float* __restrict__ g,
                                     const float* __restrict__ b,
                                     _Float16* __restrict__ Y, int ldy,
                                     int M, int D, int relu)
{
    const int row = blockIdx.x * blockDim.y + threadIdx.y;
    if (row >= M) return;
    const int lane = threadIdx.x;
    const float* x = X + (long)row * ldx;

    float s = 0.0f, ss = 0.0f;
    for (int i = lane; i < D; i += 32) { float v = x[i]; s += v; ss += v * v; }
#pragma unroll
    for (int off = 16; off > 0; off >>= 1) {
        s  += __shfl_xor(s,  off, 32);
        ss += __shfl_xor(ss, off, 32);
    }
    const float mu  = s / (float)D;
    const float inv = rsqrtf(ss / (float)D - mu * mu + 1e-5f);

    _Float16* y = Y + (long)row * ldy;
    for (int i = lane; i < D; i += 32) {
        float v = (x[i] - mu) * inv * g[i] + b[i];
        if (relu) v = fmaxf(v, 0.0f);
        y[i] = (_Float16)v;
    }
}

// ResGated edge pass, 4 channels per thread (float4 gathers, f32 atomics):
//   out[dst] += sigmoid(k[dst]+q[src]+eW[e]) * v[src]
__global__ void edge_scatter_kernel(const int* __restrict__ ei, long E,
                                    const float* __restrict__ Kf,
                                    const float* __restrict__ Qf,
                                    const float* __restrict__ Vf,
                                    const float* __restrict__ eW,
                                    float* __restrict__ out,
                                    int D4, long e0, long eCnt)
{
    const long idx = (long)blockIdx.x * blockDim.x + threadIdx.x;
    if (idx >= eCnt * (long)D4) return;
    const long el = idx / D4;
    const int  j  = (int)(idx - el * (long)D4);
    const long e  = e0 + el;
    const int src = ei[e];
    const int dst = ei[E + e];

    const float4* K4 = (const float4*)Kf;
    const float4* Q4 = (const float4*)Qf;
    const float4* V4 = (const float4*)Vf;
    const float4* E4 = (const float4*)eW;

    const float4 kk = K4[(long)dst * D4 + j];
    const float4 qq = Q4[(long)src * D4 + j];
    const float4 vv = V4[(long)src * D4 + j];
    const float4 ee = E4[el * (long)D4 + j];

    float4 r;
    r.x = vv.x / (1.0f + __expf(-(kk.x + qq.x + ee.x)));
    r.y = vv.y / (1.0f + __expf(-(kk.y + qq.y + ee.y)));
    r.z = vv.z / (1.0f + __expf(-(kk.z + qq.z + ee.z)));
    r.w = vv.w / (1.0f + __expf(-(kk.w + qq.w + ee.w)));

    float* o = out + (long)dst * (4 * (long)D4) + 4 * (long)j;
    atomicAdd(o + 0, r.x);
    atomicAdd(o + 1, r.y);
    atomicAdd(o + 2, r.z);
    atomicAdd(o + 3, r.w);
}

// Weight prep: Wt[n*Kp+k] = (n<N && k<K) ? W[k*N+n] : 0   (f32 -> f16, T + pad)
__global__ void wt_f16_kernel(const float* __restrict__ W, _Float16* __restrict__ Wt,
                              int K, int N, int Kp, int Np)
{
    const long idx = (long)blockIdx.x * blockDim.x + threadIdx.x;
    if (idx >= (long)Np * Kp) return;
    const int n = (int)(idx / Kp);
    const int k = (int)(idx - (long)n * Kp);
    Wt[idx] = (n < N && k < K) ? (_Float16)W[(long)k * N + n] : (_Float16)0.0f;
}

// Activation pad: Y[i*Kp+c] = c<K ? (f16)X[i*K+c] : 0
__global__ void pad_f16_kernel(const float* __restrict__ X, int K,
                               _Float16* __restrict__ Y, int Kp, long M)
{
    const long idx = (long)blockIdx.x * blockDim.x + threadIdx.x;
    if (idx >= M * (long)Kp) return;
    const long i = idx / Kp;
    const int  c = (int)(idx - i * Kp);
    Y[idx] = (c < K) ? (_Float16)X[i * (long)K + c] : (_Float16)0.0f;
}

// IDG encoder input, padded f16: [derived(29) | emb[type](8) | 0...] -> [N,64]
__global__ void build_idgin_kernel(const float* __restrict__ der,
                                   const int* __restrict__ typ,
                                   const float* __restrict__ emb,
                                   _Float16* __restrict__ out, int n)
{
    const long idx = (long)blockIdx.x * blockDim.x + threadIdx.x;
    if (idx >= (long)n * 64) return;
    const int i = (int)(idx >> 6);
    const int c = (int)(idx & 63);
    float v = 0.0f;
    if (c < 29)      v = der[(long)i * 29 + c];
    else if (c < 37) v = emb[(long)typ[i] * 8 + (c - 29)];
    out[idx] = (_Float16)v;
}

// combined = [idg_feat(128,f32) | featA[idxA](64,f16) | featB[idxB](64,f16)] -> f16 [N,256]
__global__ void build_combined_kernel(const float* __restrict__ idgf,
                                      const _Float16* __restrict__ fA,
                                      const _Float16* __restrict__ fB,
                                      const int* __restrict__ iA,
                                      const int* __restrict__ iB,
                                      _Float16* __restrict__ comb, int n)
{
    const long idx = (long)blockIdx.x * blockDim.x + threadIdx.x;
    if (idx >= (long)n * 256) return;
    const int i = (int)(idx >> 8);
    const int c = (int)(idx & 255);
    float v;
    if (c < 128)      v = idgf[(long)i * 128 + c];
    else if (c < 192) v = (float)fA[(long)iA[i] * 64 + (c - 128)];
    else              v = (float)fB[(long)iB[i] * 64 + (c - 192)];
    comb[idx] = (_Float16)v;
}

__global__ void softmax_stats_kernel(const float* __restrict__ s, int n,
                                     float* __restrict__ stats)
{
    __shared__ float sm[512];
    const int t = threadIdx.x;
    float m = -1e30f;
    for (int i = t; i < n; i += blockDim.x) m = fmaxf(m, s[i]);
    sm[t] = m; __syncthreads();
    for (int off = blockDim.x >> 1; off > 0; off >>= 1) {
        if (t < off) sm[t] = fmaxf(sm[t], sm[t + off]);
        __syncthreads();
    }
    const float mx = sm[0]; __syncthreads();
    float z = 0.0f;
    for (int i = t; i < n; i += blockDim.x) z += __expf(s[i] - mx);
    sm[t] = z; __syncthreads();
    for (int off = blockDim.x >> 1; off > 0; off >>= 1) {
        if (t < off) sm[t] += sm[t + off];
        __syncthreads();
    }
    if (t == 0) { stats[0] = mx; stats[1] = sm[0]; }
}

__global__ void attn_pool_kernel(const float* __restrict__ post,
                                 const float* __restrict__ s,
                                 const float* __restrict__ stats,
                                 float* __restrict__ pooled, int n, int D)
{
    const int c = blockIdx.x * blockDim.x + threadIdx.x;
    if (c >= D) return;
    const float mx = stats[0], z = stats[1];
    float acc = 0.0f;
    for (int i = 0; i < n; ++i)
        acc += post[(long)i * D + c] * __expf(s[i] - mx);
    pooled[c] = acc / z;
}

__global__ void final_kernel(const float* __restrict__ pooled,
                             const float* __restrict__ w,
                             const float* __restrict__ b,
                             float* __restrict__ out, int D)
{
    __shared__ float sm[256];
    const int t = threadIdx.x;
    sm[t] = (t < D) ? pooled[t] * w[t] : 0.0f;
    __syncthreads();
    for (int off = 128; off > 0; off >>= 1) {
        if (t < off) sm[t] += sm[t + off];
        __syncthreads();
    }
    if (t == 0) out[0] = 1.0f / (1.0f + __expf(-(sm[0] + b[0])));
}

// ---------------------------------------------------------------------------
// Host orchestration
// ---------------------------------------------------------------------------
struct LinP  { const float* w; const float* b; };
struct ConvP { LinP edge, key, query, skip, value; };
struct BlockP {
    ConvP c[3];
    LinP lin_edge, lin_hidden, lin_node;
    const float* ln_b[3];
    const float* ln_g[3];
};
struct WT { const _Float16* t; int Kp; int Np; };
struct BlockW {
    WT c_edge[3], c_key[3], c_query[3], c_skip[3], c_value[3];
    WT lin_edge, lin_hidden, lin_node;
};

extern "C" void kernel_launch(void* const* d_in, const int* in_sizes, int n_in,
                              void* d_out, int out_size, void* d_ws, size_t ws_size,
                              hipStream_t stream)
{
    (void)n_in; (void)out_size; (void)ws_size;

    auto F = [&](int i) { return (const float*)d_in[i]; };

    const float* nodes_A = F(0);
    const int*   eiA     = (const int*)d_in[1];
    const float* eaA     = F(2);
    const float* nodes_B = F(3);
    const int*   eiB     = (const int*)d_in[4];
    const float* eaB     = F(5);
    const float* idg_der = F(6);
    const int*   idg_typ = (const int*)d_in[7];
    const int*   eiI     = (const int*)d_in[8];
    const float* eaI     = F(9);
    const int*   idxA    = (const int*)d_in[10];
    const int*   idxB    = (const int*)d_in[11];

    const long N_A   = in_sizes[0] / 41;
    const long E_A   = in_sizes[1] / 2;
    const long N_B   = in_sizes[3] / 41;
    const long E_B   = in_sizes[4] / 2;
    const long N_IDG = in_sizes[6] / 29;
    const long E_IDG = in_sizes[8] / 2;

    // ---- params (JAX pytree: alphabetically sorted dict keys) ----
    int pi = 12;
    auto lin = [&]() { LinP p; p.b = F(pi); p.w = F(pi + 1); pi += 2; return p; };
    LinP attn1 = lin();
    LinP attn2 = lin();
    const float* emb = F(pi++);
    LinP enc1 = lin();
    LinP enc2 = lin();
    const float* enc_ln_b = F(pi++);
    const float* enc_ln_g = F(pi++);
    LinP finp = lin();
    auto conv = [&]() {
        ConvP cc;
        cc.edge = lin(); cc.key = lin(); cc.query = lin(); cc.skip = lin(); cc.value = lin();
        return cc;
    };
    auto block = [&]() {
        BlockP bb;
        for (int i = 0; i < 3; ++i) bb.c[i] = conv();
        bb.lin_edge = lin(); bb.lin_hidden = lin(); bb.lin_node = lin();
        for (int i = 0; i < 3; ++i) { bb.ln_b[i] = F(pi++); bb.ln_g[i] = F(pi++); }
        return bb;
    };
    BlockP mono  = block();
    BlockP postb = block();

    // ---- workspace carve (64B-aligned) ----
    char* wsb = (char*)d_ws;
    size_t off = 0;
    auto allocB = [&](size_t bytes) {
        void* p = wsb + off;
        off = (off + bytes + 63) & ~(size_t)63;
        return p;
    };
    auto allocF = [&](size_t n) { return (float*)allocB(n * 4); };
    auto allocH = [&](size_t n) { return (_Float16*)allocB(n * 2); };
    const long CHUNK = 131072;
    const long Emax  = (E_A > E_IDG) ? E_A : E_IDG;
    const size_t nodeMax = (size_t)((N_A * 64 > N_IDG * 256) ? N_A * 64 : N_IDG * 256);

    _Float16* idg_in_h   = allocH((size_t)N_IDG * 64);
    float*    idg_h      = allocF((size_t)N_IDG * 128);
    _Float16* idg_h_h    = allocH((size_t)N_IDG * 128);
    float*    idg_feat   = allocF((size_t)N_IDG * 128);
    _Float16* nodes_h    = allocH((size_t)N_A * 64);
    _Float16* eattr_h    = allocH((size_t)Emax * 32);
    _Float16* ef_h       = allocH((size_t)Emax * 32);
    _Float16* hbuf_h     = allocH(nodeMax);
    _Float16* hs_h       = allocH(3 * nodeMax);
    float*    kbuf       = allocF(nodeMax);
    float*    qbuf       = allocF(nodeMax);
    float*    vbuf       = allocF(nodeMax);
    float*    convout    = allocF(nodeMax);
    float*    eWbuf      = allocF((size_t)CHUNK * 256);
    _Float16* featA_h    = allocH((size_t)N_A * 64);
    _Float16* featB_h    = allocH((size_t)N_B * 64);
    _Float16* combined_h = allocH((size_t)N_IDG * 256);
    float*    post_out   = allocF((size_t)N_IDG * 256);
    _Float16* post_out_h = allocH((size_t)N_IDG * 256);
    _Float16* attnh_h    = allocH((size_t)N_IDG * 128);
    float*    scores     = allocF((size_t)N_IDG);
    float*    stats      = allocF(8);
    float*    pooled     = allocF(256);

    // ---- weight prep: transpose + pad to f16 [Np, Kp] ----
    auto prep_w = [&](LinP p, int K, int N) -> WT {
        const int Kp = (K + 31) & ~31;
        const int Np = (N + 63) & ~63;
        _Float16* t = allocH((size_t)Np * Kp);
        const long tot = (long)Np * Kp;
        wt_f16_kernel<<<(unsigned)((tot + 255) / 256), 256, 0, stream>>>(p.w, t, K, N, Kp, Np);
        return WT{t, Kp, Np};
    };
    auto prep_block = [&](const BlockP& bp, int inDim, int eIn, int D) -> BlockW {
        BlockW w;
        for (int i = 0; i < 3; ++i) {
            w.c_edge[i]  = prep_w(bp.c[i].edge,  32, D);
            w.c_key[i]   = prep_w(bp.c[i].key,    D, D);
            w.c_query[i] = prep_w(bp.c[i].query,  D, D);
            w.c_skip[i]  = prep_w(bp.c[i].skip,   D, D);
            w.c_value[i] = prep_w(bp.c[i].value,  D, D);
        }
        w.lin_edge   = prep_w(bp.lin_edge,   eIn,   32);
        w.lin_hidden = prep_w(bp.lin_hidden, 3 * D, D);
        w.lin_node   = prep_w(bp.lin_node,   inDim, D);
        return w;
    };
    WT w_attn1 = prep_w(attn1, 256, 128);
    WT w_attn2 = prep_w(attn2, 128, 1);
    WT w_enc1  = prep_w(enc1,  37, 128);
    WT w_enc2  = prep_w(enc2, 128, 128);
    BlockW mw = prep_block(mono,  41, 21,  64);
    BlockW pw = prep_block(postb, 256, 26, 256);

    // ---- launch helpers ----
    auto gemm = [&](const _Float16* A, int lda, WT w, const float* bias,
                    float* Yf, _Float16* Yh, int ldy, long M, int N, int act) {
        dim3 blk(32, 4);
        const unsigned tilesM = (unsigned)(M / 16);
        dim3 grd((unsigned)(w.Np / 64), (tilesM + 3) / 4);
        wmma_gemm_kernel<<<grd, blk, 0, stream>>>(A, lda, w.t, w.Kp, bias,
                                                  Yf, Yh, ldy, (int)M, N, act);
    };
    auto lnorm = [&](const float* X, int ldx, const float* g, const float* b,
                     _Float16* Y, int ldy, long M, int D, int relu) {
        dim3 blk(32, 4);
        layernorm_f16_kernel<<<(unsigned)((M + 3) / 4), blk, 0, stream>>>(
            X, ldx, g, b, Y, ldy, (int)M, D, relu);
    };
    auto pad16 = [&](const float* X, int K, _Float16* Y, int Kp, long M) {
        const long tot = M * (long)Kp;
        pad_f16_kernel<<<(unsigned)((tot + 255) / 256), 256, 0, stream>>>(X, K, Y, Kp, M);
    };

    auto run_block = [&](const BlockW& bw, const BlockP& bp,
                         const _Float16* x_h, int lda, long Nn,
                         const int* ei, long E, const float* eattr, int eIn,
                         int D, _Float16* out_h, float* out_f) {
        gemm(x_h, lda, bw.lin_node, bp.lin_node.b, nullptr, hbuf_h, D, Nn, D, 0);
        pad16(eattr, eIn, eattr_h, bw.lin_edge.Kp, E);
        gemm(eattr_h, bw.lin_edge.Kp, bw.lin_edge, bp.lin_edge.b, nullptr, ef_h, 32, E, 32, 0);
        for (int i = 0; i < 3; ++i) {
            const _Float16* xin = (i == 0) ? hbuf_h : (hs_h + (size_t)(i - 1) * D);
            const int       lx  = (i == 0) ? D : 3 * D;
            gemm(xin, lx, bw.c_skip[i],  bp.c[i].skip.b,  convout, nullptr, D, Nn, D, 0);
            gemm(xin, lx, bw.c_key[i],   bp.c[i].key.b,   kbuf,    nullptr, D, Nn, D, 0);
            gemm(xin, lx, bw.c_query[i], bp.c[i].query.b, qbuf,    nullptr, D, Nn, D, 0);
            gemm(xin, lx, bw.c_value[i], bp.c[i].value.b, vbuf,    nullptr, D, Nn, D, 0);
            const int D4 = D / 4;
            for (long e0 = 0; e0 < E; e0 += CHUNK) {
                const long ec = (E - e0 < CHUNK) ? (E - e0) : CHUNK;
                gemm(ef_h + (size_t)e0 * 32, 32, bw.c_edge[i], bp.c[i].edge.b,
                     eWbuf, nullptr, D, ec, D, 0);
                const long tot = ec * (long)D4;
                edge_scatter_kernel<<<(unsigned)((tot + 255) / 256), 256, 0, stream>>>(
                    ei, E, kbuf, qbuf, vbuf, eWbuf, convout, D4, e0, ec);
            }
            lnorm(convout, D, bp.ln_g[i], bp.ln_b[i], hs_h + (size_t)i * D, 3 * D, Nn, D, 0);
        }
        gemm(hs_h, 3 * D, bw.lin_hidden, bp.lin_hidden.b, out_f, out_h, D, Nn, D, 0);
    };

    // ---- 1. IDG node encoder ----
    {
        const long tot = N_IDG * 64;
        build_idgin_kernel<<<(unsigned)((tot + 255) / 256), 256, 0, stream>>>(
            idg_der, idg_typ, emb, idg_in_h, (int)N_IDG);
        gemm(idg_in_h, 64, w_enc1, enc1.b, idg_h, nullptr, 128, N_IDG, 128, 0);
        lnorm(idg_h, 128, enc_ln_g, enc_ln_b, idg_h_h, 128, N_IDG, 128, /*relu=*/1);
        gemm(idg_h_h, 128, w_enc2, enc2.b, idg_feat, nullptr, 128, N_IDG, 128, 0);
    }

    // ---- 2. twin monomer GNN stacks (shared weights) ----
    pad16(nodes_A, 41, nodes_h, mw.lin_node.Kp, N_A);
    run_block(mw, mono, nodes_h, mw.lin_node.Kp, N_A, eiA, E_A, eaA, 21, 64, featA_h, nullptr);
    pad16(nodes_B, 41, nodes_h, mw.lin_node.Kp, N_B);
    run_block(mw, mono, nodes_h, mw.lin_node.Kp, N_B, eiB, E_B, eaB, 21, 64, featB_h, nullptr);

    // ---- 3. combine + post GNN block ----
    {
        const long tot = N_IDG * 256;
        build_combined_kernel<<<(unsigned)((tot + 255) / 256), 256, 0, stream>>>(
            idg_feat, featA_h, featB_h, idxA, idxB, combined_h, (int)N_IDG);
    }
    run_block(pw, postb, combined_h, 256, N_IDG, eiI, E_IDG, eaI, 26, 256, nullptr, post_out);

    // ---- 4. attention pooling + final ----
    pad16(post_out, 256, post_out_h, 256, N_IDG);
    gemm(post_out_h, 256, w_attn1, attn1.b, nullptr, attnh_h, 128, N_IDG, 128, /*tanh=*/1);
    gemm(attnh_h, 128, w_attn2, attn2.b, scores, nullptr, 1, N_IDG, 1, 0);
    softmax_stats_kernel<<<1, 512, 0, stream>>>(scores, (int)N_IDG, stats);
    attn_pool_kernel<<<1, 256, 0, stream>>>(post_out, scores, stats, pooled, (int)N_IDG, 256);
    final_kernel<<<1, 256, 0, stream>>>(pooled, finp.w, finp.b, (float*)d_out, 256);
}
